// DeepSetsInvariantFSPool_32753420599598
// MI455X (gfx1250) — matrix operand order repaired and checked by reading, not compiled, not run on hardware
//
#include <hip/hip_runtime.h>
#include <hip/hip_bf16.h>

// ---------------------------------------------------------------------------
// DeepSets + FSPool for MI455X (gfx1250, wave32, WMMA)
//   phi:   [B*S,64] x [64,256] -> relu -> x [256,256]  (bf16 WMMA, f32 acc)
//   pool:  per (b,c) bitonic sort of 1024 + piecewise-linear weights
//   rho:   [128,256] x [256,256] -> relu -> x [256,128] (bf16 WMMA)
// ---------------------------------------------------------------------------

typedef __attribute__((ext_vector_type(16))) __bf16 v16bf;
typedef __attribute__((ext_vector_type(8)))  __bf16 v8bf;
typedef __attribute__((ext_vector_type(8)))  float  v8f;
typedef __attribute__((ext_vector_type(4)))  float  v4f;

#define Bz     128
#define Sz     1024
#define DIN    64
#define DH     256
#define DLAT   256
#define DOUT   128
#define NPIECE 20

// ---------------------------------------------------------------------------
// Pack an f32 [K][N] weight matrix into bf16 WMMA B-fragments.
// Fragment (kt,nt) covers K rows [32kt,32kt+32), cols [16nt,16nt+16).
// Lane l: n = 16nt + (l&15), half = l>>4; elems j<8 -> k = 32kt + 8*half + j
//         elems j>=8 -> k = 32kt + 16 + 8*half + (j-8)
// Storage: frag[((nt*KT + kt)*32 + lane)*16 + j]  (32B per lane, coalesced)
// ---------------------------------------------------------------------------
__global__ void prep_frags(const float* __restrict__ W, __bf16* __restrict__ frag,
                           int K, int N) {
  int KT = K >> 5;
  int total = (K >> 5) * (N >> 4) * 32;
  int tid = blockIdx.x * blockDim.x + threadIdx.x;
  if (tid >= total) return;
  int lane = tid & 31;
  int fid  = tid >> 5;
  int nt = fid / KT, kt = fid % KT;
  int n = nt * 16 + (lane & 15);
  int half = lane >> 4;
  __bf16* dst = frag + (size_t)tid * 16;
#pragma unroll
  for (int j = 0; j < 16; ++j) {
    int k = kt * 32 + (j < 8 ? 8 * half + j : 16 + 8 * half + (j - 8));
    dst[j] = (__bf16)W[(size_t)k * N + n];
  }
}

// ---------------------------------------------------------------------------
// n[b] = sum_s mask[b,s]
// ---------------------------------------------------------------------------
__global__ void count_kernel(const int* __restrict__ mask, int* __restrict__ nvals) {
  __shared__ int red[256];
  int b = blockIdx.x, t = threadIdx.x;
  int acc = 0;
  for (int s = t; s < Sz; s += 256) acc += mask[b * Sz + s];
  red[t] = acc;
  __syncthreads();
  for (int off = 128; off > 0; off >>= 1) {
    if (t < off) red[t] += red[t + off];
    __syncthreads();
  }
  if (t == 0) nvals[b] = red[0];
}

// Build a 16x32 bf16 A-fragment from an f32 row pointer (per-lane row).
// Lane (half = lane>>4): elems 0..7  -> k = 8*half + j
//                        elems 8..15 -> k = 16 + 8*half + j
__device__ __forceinline__ v16bf load_a_frag_f32(const float* p, int half) {
  v4f a0 = *(const v4f*)(p + 8 * half);
  v4f a1 = *(const v4f*)(p + 8 * half + 4);
  v4f a2 = *(const v4f*)(p + 16 + 8 * half);
  v4f a3 = *(const v4f*)(p + 16 + 8 * half + 4);
  v16bf av;
#pragma unroll
  for (int j = 0; j < 4; ++j) {
    av[j]      = (__bf16)a0[j];
    av[j + 4]  = (__bf16)a1[j];
    av[j + 8]  = (__bf16)a2[j];
    av[j + 12] = (__bf16)a3[j];
  }
  return av;
}

// ---------------------------------------------------------------------------
// Fused phi MLP: feat[b][c][s] = (relu(x W1 + b1) W2 + b2)  (transposed store)
// 4 waves/block, each wave owns 16 rows; h tile staged in LDS as bf16.
// ---------------------------------------------------------------------------
__global__ void __launch_bounds__(128) phi_kernel(
    const float* __restrict__ x, const float* __restrict__ b1,
    const float* __restrict__ b2, const __bf16* __restrict__ w1f,
    const __bf16* __restrict__ w2f, float* __restrict__ feat) {
  __shared__ __bf16 hbuf[4][16][DH + 8];   // +8 bf16 pad -> 528B rows, bank rotate

  int blk  = blockIdx.x;
  int b    = blk >> 4;          // 16 s-tiles of 64 rows
  int tile = blk & 15;
  int wave = threadIdx.x >> 5;
  int lane = threadIdx.x & 31;
  int half = lane >> 4;
  int mrow = lane & 15;
  int ncol = lane & 15;
  int srow = tile * 64 + wave * 16;

  const float* xrow = x + (size_t)(b * Sz + srow + mrow) * DIN;

  // ---- layer 1: [16x64] x [64x256] + b1, relu -> LDS bf16 ----
#pragma unroll 4
  for (int nt = 0; nt < 16; ++nt) {
    v8f acc = {};
#pragma unroll
    for (int kt = 0; kt < 2; ++kt) {
      v16bf av = load_a_frag_f32(xrow + kt * 32, half);
      v16bf bv = *(const v16bf*)(w1f + ((size_t)(nt * 2 + kt) * 32 + lane) * 16);
      acc = __builtin_amdgcn_wmma_f32_16x16x32_bf16(false, av, false, bv,
                                                    (short)0, acc, false, false);
    }
    float bias = b1[nt * 16 + ncol];
#pragma unroll
    for (int r = 0; r < 8; ++r) {
      float h = acc[r] + bias;
      h = h > 0.0f ? h : 0.0f;
      hbuf[wave][r + 8 * half][nt * 16 + ncol] = (__bf16)h;
    }
  }
  __syncthreads();

  // ---- layer 2: [16x256] x [256x256] + b2 -> feat[b][c][s] ----
#pragma unroll 2
  for (int nt = 0; nt < 16; ++nt) {
    v8f acc = {};
#pragma unroll
    for (int kt = 0; kt < 8; ++kt) {
      v8bf lo = *(const v8bf*)&hbuf[wave][mrow][kt * 32 + 8 * half];
      v8bf hi = *(const v8bf*)&hbuf[wave][mrow][kt * 32 + 16 + 8 * half];
      v16bf av;
#pragma unroll
      for (int j = 0; j < 8; ++j) { av[j] = lo[j]; av[j + 8] = hi[j]; }
      v16bf bv = *(const v16bf*)(w2f + ((size_t)(nt * 8 + kt) * 32 + lane) * 16);
      acc = __builtin_amdgcn_wmma_f32_16x16x32_bf16(false, av, false, bv,
                                                    (short)0, acc, false, false);
    }
    int c = nt * 16 + ncol;
    float bias = b2[c];
    float* op = feat + ((size_t)b * DLAT + c) * Sz + srow + 8 * half;
    v4f o0 = {acc[0] + bias, acc[1] + bias, acc[2] + bias, acc[3] + bias};
    v4f o1 = {acc[4] + bias, acc[5] + bias, acc[6] + bias, acc[7] + bias};
    *(v4f*)op = o0;
    *(v4f*)(op + 4) = o1;
  }
}

// ---------------------------------------------------------------------------
// FSPool: one block per (b,c). Bitonic descending sort of 1024 floats in LDS,
// then piecewise-linear-weighted masked sum -> pooled[b][c].
// ---------------------------------------------------------------------------
__global__ void __launch_bounds__(512) fspool_kernel(
    const float* __restrict__ feat, const int* __restrict__ nvals,
    const float* __restrict__ pw, float* __restrict__ pooled) {
  __shared__ float sd[Sz];
  __shared__ float red[512];
  int b = blockIdx.x >> 8;
  int c = blockIdx.x & 255;
  int t = threadIdx.x;

  int nb = nvals[b];
  float denom = fmaxf((float)nb - 1.0f, 1.0f);
  const float* fr = feat + ((size_t)b * DLAT + c) * Sz;
  __builtin_prefetch(fr, 0, 1);            // global_prefetch into WGP$/GL2

  for (int i = t; i < Sz; i += 512) {
    float ratio = (float)i / denom;
    float m = (ratio <= 1.0f) ? 1.0f : 0.0f;
    sd[i] = fr[i] + (1.0f - m) * (-99999.0f);
  }
  __syncthreads();

  // bitonic sort, descending
  for (int k = 2; k <= Sz; k <<= 1) {
    for (int j = k >> 1; j > 0; j >>= 1) {
      for (int i = t; i < Sz; i += 512) {
        int ixj = i ^ j;
        if (ixj > i) {
          float a0 = sd[i], a1 = sd[ixj];
          bool up = (i & k) == 0;                 // "up" blocks: descending
          bool sw = up ? (a0 < a1) : (a0 > a1);
          if (sw) { sd[i] = a1; sd[ixj] = a0; }
        }
      }
      __syncthreads();
    }
  }

  // weighted masked sum over sorted values
  float acc = 0.0f;
  const float* wr = pw + c * (NPIECE + 1);
  for (int i = t; i < Sz; i += 512) {
    float ratio = (float)i / denom;
    float m = (ratio <= 1.0f) ? 1.0f : 0.0f;
    ratio = fminf(ratio, 1.0f);
    float pos = (float)NPIECE * ratio;
    int idx = (int)floorf(pos);
    if (idx > NPIECE) idx = NPIECE;
    float frac = pos - (float)idx;
    int idx1 = idx + 1;
    if (idx1 > NPIECE) idx1 = NPIECE;
    float w = (1.0f - frac) * wr[idx] + frac * wr[idx1];
    acc += sd[i] * w * m;
  }
  red[t] = acc;
  __syncthreads();
  for (int off = 256; off > 0; off >>= 1) {
    if (t < off) red[t] += red[t + off];
    __syncthreads();
  }
  if (t == 0) pooled[b * DLAT + c] = red[0];
}

// ---------------------------------------------------------------------------
// rho MLP: out = relu(pooled W1 + b1) W2 + b2  ([128,256]->[128,256]->[128,128])
// 2 blocks x 4 waves; same fused-in-LDS scheme as phi.
// ---------------------------------------------------------------------------
__global__ void __launch_bounds__(128) rho_kernel(
    const float* __restrict__ pooled, const float* __restrict__ b1,
    const float* __restrict__ b2, const __bf16* __restrict__ w1f,
    const __bf16* __restrict__ w2f, float* __restrict__ out) {
  __shared__ __bf16 hbuf[4][16][DH + 8];
  int wave = threadIdx.x >> 5;
  int lane = threadIdx.x & 31;
  int half = lane >> 4;
  int mrow = lane & 15;
  int ncol = lane & 15;
  int rbase = blockIdx.x * 64 + wave * 16;

  const float* prow = pooled + (size_t)(rbase + mrow) * DLAT;

  // layer 1: [16x256] x [256x256] + b1, relu -> LDS
  for (int nt = 0; nt < 16; ++nt) {
    v8f acc = {};
#pragma unroll
    for (int kt = 0; kt < 8; ++kt) {
      v16bf av = load_a_frag_f32(prow + kt * 32, half);
      v16bf bv = *(const v16bf*)(w1f + ((size_t)(nt * 8 + kt) * 32 + lane) * 16);
      acc = __builtin_amdgcn_wmma_f32_16x16x32_bf16(false, av, false, bv,
                                                    (short)0, acc, false, false);
    }
    float bias = b1[nt * 16 + ncol];
#pragma unroll
    for (int r = 0; r < 8; ++r) {
      float h = acc[r] + bias;
      h = h > 0.0f ? h : 0.0f;
      hbuf[wave][r + 8 * half][nt * 16 + ncol] = (__bf16)h;
    }
  }
  __syncthreads();

  // layer 2: [16x256] x [256x128] + b2 -> out
  for (int nt = 0; nt < 8; ++nt) {
    v8f acc = {};
#pragma unroll
    for (int kt = 0; kt < 8; ++kt) {
      v8bf lo = *(const v8bf*)&hbuf[wave][mrow][kt * 32 + 8 * half];
      v8bf hi = *(const v8bf*)&hbuf[wave][mrow][kt * 32 + 16 + 8 * half];
      v16bf av;
#pragma unroll
      for (int j = 0; j < 8; ++j) { av[j] = lo[j]; av[j + 8] = hi[j]; }
      v16bf bv = *(const v16bf*)(w2f + ((size_t)(nt * 8 + kt) * 32 + lane) * 16);
      acc = __builtin_amdgcn_wmma_f32_16x16x32_bf16(false, av, false, bv,
                                                    (short)0, acc, false, false);
    }
    int n = nt * 16 + ncol;
    float bias = b2[n];
#pragma unroll
    for (int r = 0; r < 8; ++r) {
      int m = rbase + r + 8 * half;
      out[(size_t)m * DOUT + n] = acc[r] + bias;
    }
  }
}

// ---------------------------------------------------------------------------
// Workspace layout (bytes):
//   nvals   @ 0       : 128 * 4                      (pad to 1024)
//   phi1f   @ 1024    : 64*256  bf16 =  32768
//   phi2f   @ 33792   : 256*256 bf16 = 131072
//   rho1f   @ 164864  : 256*256 bf16 = 131072
//   rho2f   @ 295936  : 256*128 bf16 =  65536
//   pooled  @ 361472  : 128*256 f32  = 131072
//   feat    @ 492544  : 128*256*1024 f32 = 134217728
// ---------------------------------------------------------------------------
extern "C" void kernel_launch(void* const* d_in, const int* in_sizes, int n_in,
                              void* d_out, int out_size, void* d_ws, size_t ws_size,
                              hipStream_t stream) {
  const float* x      = (const float*)d_in[0];
  const int*   mask   = (const int*)d_in[1];
  const float* phi_w1 = (const float*)d_in[2];
  const float* phi_b1 = (const float*)d_in[3];
  const float* phi_w2 = (const float*)d_in[4];
  const float* phi_b2 = (const float*)d_in[5];
  const float* rho_w1 = (const float*)d_in[6];
  const float* rho_b1 = (const float*)d_in[7];
  const float* rho_w2 = (const float*)d_in[8];
  const float* rho_b2 = (const float*)d_in[9];
  const float* pool_w = (const float*)d_in[10];

  char* ws = (char*)d_ws;
  int*    nvals  = (int*)(ws + 0);
  __bf16* phi1f  = (__bf16*)(ws + 1024);
  __bf16* phi2f  = (__bf16*)(ws + 33792);
  __bf16* rho1f  = (__bf16*)(ws + 164864);
  __bf16* rho2f  = (__bf16*)(ws + 295936);
  float*  pooled = (float*)(ws + 361472);
  float*  feat   = (float*)(ws + 492544);

  // pack weights into WMMA B-fragment order (bf16)
  prep_frags<<<dim3((2 * 16 * 32 + 255) / 256), 256, 0, stream>>>(phi_w1, phi1f, DIN, DH);
  prep_frags<<<dim3((8 * 16 * 32 + 255) / 256), 256, 0, stream>>>(phi_w2, phi2f, DH, DLAT);
  prep_frags<<<dim3((8 * 16 * 32 + 255) / 256), 256, 0, stream>>>(rho_w1, rho1f, DLAT, DH);
  prep_frags<<<dim3((8 * 8 * 32 + 255) / 256), 256, 0, stream>>>(rho_w2, rho2f, DH, DOUT);

  count_kernel<<<dim3(Bz), 256, 0, stream>>>(mask, nvals);

  phi_kernel<<<dim3(Bz * (Sz / 64)), 128, 0, stream>>>(x, phi_b1, phi_b2,
                                                       phi1f, phi2f, feat);

  fspool_kernel<<<dim3(Bz * DLAT), 512, 0, stream>>>(feat, nvals, pool_w, pooled);

  rho_kernel<<<dim3(2), 128, 0, stream>>>(pooled, rho_b1, rho_b2,
                                          rho1f, rho2f, (float*)d_out);
}
